// GraphTransformerLayer_16286515986914
// MI455X (gfx1250) — matrix operand order repaired and checked by reading, not compiled, not run on hardware
//
#include <hip/hip_runtime.h>
#include <math.h>

#define NN 100000
#define EE 1600000
#define CC 64
#define HH 8
#define DD 8
#define C2 128
#define LN_EPS 1e-5f

typedef __attribute__((ext_vector_type(16))) _Float16 v16h;
typedef __attribute__((ext_vector_type(8)))  float    v8f;
typedef _Float16 half_t;

__device__ __forceinline__ v8f wmma_f16(v16h a, v16h b, v8f c) {
  // D = A(16x32 f16) * B(32x16 f16) + C(16x16 f32)
  return __builtin_amdgcn_wmma_f32_16x16x32_f16(false, a, false, b, (short)0, c,
                                                false, false);
}

// ---- fragment-packed layouts ------------------------------------------------
// A-fragment ISA layout (16x32 f16): lanes 0-15: M=lane, K 0..7 / 16..23;
// lanes 16-31: M=lane-16, K 8..15 / 24..31. We store A already packed so each
// lane reads 16 contiguous f16 (2x ds_load_b128).
__device__ __forceinline__ void pack_a_store(half_t* buf /*[kc][32][16]*/,
                                             int m, int k, float val) {
  int kc = k >> 5, kk = k & 31;
  int group = kk >> 3, rem = kk & 7;
  int lane = m + ((group & 1) ? 16 : 0);
  int i    = rem + ((group & 2) ? 8 : 0);
  buf[(kc * 32 + lane) * 16 + i] = (half_t)val;
}

// B-fragments prepacked in global ws: [nt][kc][lane][16] f16, contiguous per lane.
__device__ __forceinline__ v16h frag_bp(const half_t* pw, int Kc, int nt,
                                        int kc, int lane) {
  return *(const v16h*)(pw + (((nt * Kc) + kc) * 32 + lane) * 16);
}

// legacy A-fragment gather from f32 global (used only for agg tiles)
__device__ __forceinline__ v16h frag_a(const float* A, int lane, int kc, int lda) {
  int m  = lane & 15;
  int kb = (lane < 16) ? 0 : 8;
  v16h a;
#pragma unroll
  for (int i = 0; i < 16; ++i) {
    int k = kc * 32 + ((i < 8) ? (kb + i) : (16 + kb + (i - 8)));
    a[i] = (_Float16)A[m * lda + k];
  }
  return a;
}

// monotone float <-> uint key (atomic max on floats via u32 atomicMax)
__device__ __forceinline__ unsigned int fkey(float f) {
  unsigned int u = __float_as_uint(f);
  return (u & 0x80000000u) ? ~u : (u | 0x80000000u);
}
__device__ __forceinline__ float funkey(unsigned int k) {
  return (k & 0x80000000u) ? __uint_as_float(k & 0x7FFFFFFFu)
                           : __uint_as_float(~k);
}

// ---------------------------------------------------------------- weight pack
// W is K x Nout (row major). Output order: [nt][kc][lane][i], i contiguous.
__global__ void k_pack_w(const float* __restrict__ W, half_t* __restrict__ pw,
                         int K, int Nout) {
  int idx = blockIdx.x * 256 + threadIdx.x;
  if (idx >= K * Nout) return;
  int i    = idx & 15;
  int lane = (idx >> 4) & 31;
  int t    = idx >> 9;            // nt * Kc + kc
  int Kc   = K >> 5;
  int kc   = t % Kc;
  int nt   = t / Kc;
  int k = kc * 32 + ((lane < 16) ? 0 : 8) + ((i < 8) ? i : (i + 8));   // matches B-frag ISA layout
  int n = nt * 16 + (lane & 15);
  pw[idx] = (half_t)W[k * Nout + n];
}

// ---------------------------------------------------------------- init
__global__ void k_init(float* agg, float* denom, unsigned int* segmax) {
  long long i = (long long)blockIdx.x * blockDim.x + threadIdx.x;
  if (i < (long long)NN * CC) agg[i] = 0.0f;
  if (i < (long long)NN * HH) { denom[i] = 0.0f; segmax[i] = 0u; }
}

// ---------------------------------------------- node LN + q/k/v (WMMA)
__global__ __launch_bounds__(128) void k_node_pre(
    const float* __restrict__ node_states,
    const float* __restrict__ n1g, const float* __restrict__ n1b,
    const half_t* __restrict__ pwq, const float* __restrict__ bq,
    const half_t* __restrict__ pwk, const float* __restrict__ bk,
    const half_t* __restrict__ pwv, const float* __restrict__ bv,
    float* __restrict__ ni, float* __restrict__ qb,
    float* __restrict__ kb, float* __restrict__ vb) {
  __shared__ alignas(32) half_t ln16[4][2 * 32 * 16];
  int lane = threadIdx.x & 31;
  int w    = threadIdx.x >> 5;
  int tile = blockIdx.x * 4 + w;
  bool active = tile < (NN / 16);
  if (active && lane < 16) {
    long long row = (long long)tile * 16 + lane;
    const float* xr = node_states + row * CC;
    float s = 0.f, ss = 0.f;
    for (int j = 0; j < CC; ++j) { float x = xr[j]; s += x; ss += x * x; }
    float mean = s * (1.0f / CC);
    float rstd = rsqrtf(ss * (1.0f / CC) - mean * mean + LN_EPS);
    float* nir = ni + row * CC;
    for (int j = 0; j < CC; ++j) {
      float y = (xr[j] - mean) * rstd * n1g[j] + n1b[j];
      nir[j] = y;
      pack_a_store(ln16[w], lane, j, y);
    }
  }
  __syncthreads();
  if (!active) return;
  v16h a0 = *(const v16h*)&ln16[w][(0 * 32 + lane) * 16];
  v16h a1 = *(const v16h*)&ln16[w][(1 * 32 + lane) * 16];
  const half_t* PW[3] = { pwq, pwk, pwv };
  const float*  BB[3] = { bq, bk, bv };
  float*        OO[3] = { qb, kb, vb };
  int n  = lane & 15;
  int mb = (lane < 16) ? 0 : 8;
#pragma unroll
  for (int mtx = 0; mtx < 3; ++mtx) {
    for (int nt = 0; nt < 4; ++nt) {
      v8f acc = {};
      acc = wmma_f16(a0, frag_bp(PW[mtx], 2, nt, 0, lane), acc);
      acc = wmma_f16(a1, frag_bp(PW[mtx], 2, nt, 1, lane), acc);
      float bias = BB[mtx][nt * 16 + n];
#pragma unroll
      for (int r = 0; r < 8; ++r) {
        long long row = (long long)tile * 16 + mb + r;
        OO[mtx][row * CC + nt * 16 + n] = acc[r] + bias;
      }
    }
  }
}

// ------ edge: async stage -> LN -> es=ei@we (WMMA) -> scores + seg-max
__global__ __launch_bounds__(128) void k_edge_score(
    const float* __restrict__ edge_states,
    const float* __restrict__ e1g, const float* __restrict__ e1b,
    const half_t* __restrict__ pwe, const float* __restrict__ be,
    const int* __restrict__ edge_index,
    const float* __restrict__ qb, const float* __restrict__ kb,
    float* __restrict__ scores, unsigned int* __restrict__ segmax) {
  __shared__ alignas(32) float  raw[4][16 * CC];   // staged tile, then ei (f32)
  __shared__ alignas(32) float  esT[4][16 * CC];   // es projection (f32)
  __shared__ alignas(32) half_t eA16[4][2 * 32 * 16];
  int lane = threadIdx.x & 31;
  int w    = threadIdx.x >> 5;
  int tile = blockIdx.x * 4 + w;                   // E/16 tiles, exact
  // --- async copy 16x64 f32 tile (4KB) into LDS: 8 waves of 32x16B ---
  {
    unsigned long long g = (unsigned long long)(uintptr_t)edge_states +
                           (unsigned long long)tile * (16ull * CC * 4ull) +
                           (unsigned)(lane * 16);
    unsigned l = (unsigned)(uintptr_t)(&raw[w][0]) + (unsigned)(lane * 16);
#pragma unroll
    for (int it = 0; it < 8; ++it) {
      asm volatile("global_load_async_to_lds_b128 %0, %1, off"
                   :: "v"(l), "v"(g) : "memory");
      l += 512; g += 512;
    }
    asm volatile("s_wait_asynccnt 0x0" ::: "memory");
  }
  __syncthreads();
  if (lane < 16) {
    float* xr = raw[w] + lane * CC;
    float s = 0.f, ss = 0.f;
    for (int j = 0; j < CC; ++j) { float x = xr[j]; s += x; ss += x * x; }
    float mean = s * (1.0f / CC);
    float rstd = rsqrtf(ss * (1.0f / CC) - mean * mean + LN_EPS);
    for (int j = 0; j < CC; ++j) {
      float y = (xr[j] - mean) * rstd * e1g[j] + e1b[j];
      xr[j] = y;                      // raw now holds ei (f32), in place
      pack_a_store(eA16[w], lane, j, y);
    }
  }
  __syncthreads();
  v16h a0 = *(const v16h*)&eA16[w][(0 * 32 + lane) * 16];
  v16h a1 = *(const v16h*)&eA16[w][(1 * 32 + lane) * 16];
  int n  = lane & 15;
  int mb = (lane < 16) ? 0 : 8;
  for (int nt = 0; nt < 4; ++nt) {
    v8f acc = {};
    acc = wmma_f16(a0, frag_bp(pwe, 2, nt, 0, lane), acc);
    acc = wmma_f16(a1, frag_bp(pwe, 2, nt, 1, lane), acc);
    float bias = be[nt * 16 + n];
#pragma unroll
    for (int r = 0; r < 8; ++r)
      esT[w][(mb + r) * CC + nt * 16 + n] = acc[r] + bias;
  }
  __syncthreads();
  const float inv_sqrt_d = 0.35355339059327373f;
#pragma unroll
  for (int it = 0; it < 4; ++it) {
    int p  = it * 32 + lane;                     // (edge_local, head) 0..127
    int el = p >> 3, h = p & 7;
    long long e = (long long)tile * 16 + el;
    int s = edge_index[e];
    int d = edge_index[(long long)EE + e];
    const float* qr = qb + (long long)s * CC + h * DD;
    const float* kr = kb + (long long)d * CC + h * DD;
    const float* er = esT[w] + el * CC + h * DD;
    const float* hr = raw[w] + el * CC + h * DD;
    float sc = 0.f;
#pragma unroll
    for (int dd = 0; dd < DD; ++dd) sc += qr[dd] * kr[dd] + er[dd] * hr[dd];
    sc *= inv_sqrt_d;
    scores[e * HH + h] = sc;
    atomicMax(&segmax[(long long)s * HH + h], fkey(sc));
  }
}

// ----------------------------------------------------- exp + denom
__global__ void k_exp_denom(const int* __restrict__ edge_index,
                            float* __restrict__ scores,
                            const unsigned int* __restrict__ segmax,
                            float* __restrict__ denom) {
  long long i = (long long)blockIdx.x * blockDim.x + threadIdx.x;
  if (i >= (long long)EE * HH) return;
  long long e = i >> 3; int h = (int)(i & 7);
  int s = edge_index[e];
  float m  = funkey(segmax[(long long)s * HH + h]);
  float ex = __expf(scores[i] - m);
  scores[i] = ex;
  atomicAdd(&denom[(long long)s * HH + h], ex);
}

// ------------------------------- attention out + weighted scatter-add
__global__ void k_attn_scatter(const int* __restrict__ edge_index,
                               const float* __restrict__ scores,
                               const float* __restrict__ denom,
                               const float* __restrict__ vb,
                               const float* __restrict__ ni,
                               float* __restrict__ att_out,
                               float* __restrict__ agg) {
  long long i = (long long)blockIdx.x * blockDim.x + threadIdx.x;
  if (i >= (long long)EE * HH) return;
  long long e = i >> 3; int h = (int)(i & 7);
  int s = edge_index[e];
  int d = edge_index[(long long)EE + e];
  float a = scores[i] / (denom[(long long)s * HH + h] + 1e-12f);
  att_out[i] = a;
  const float* vr = vb + (long long)d * CC + h * DD;
  const float* nr = ni + (long long)d * CC + h * DD;
#pragma unroll
  for (int dd = 0; dd < DD; ++dd)
    atomicAdd(&agg[(long long)s * CC + h * DD + dd], a * vr[dd] * nr[dd]);
}

// -------------------- node finish: agg@wno + residual, LN, FFN (WMMA)
__global__ __launch_bounds__(128) void k_node_finish(
    const float* __restrict__ node_states, const float* __restrict__ agg,
    const half_t* __restrict__ pwno, const float* __restrict__ bno,
    const float* __restrict__ n2g, const float* __restrict__ n2b,
    const half_t* __restrict__ pnf1w, const float* __restrict__ nf1b,
    const half_t* __restrict__ pnf2w, const float* __restrict__ nf2b,
    float* __restrict__ out_node) {
  __shared__ alignas(32) float  res[4][16 * CC];
  __shared__ alignas(32) half_t xln16[4][2 * 32 * 16];
  __shared__ alignas(32) half_t h1p[4][4 * 32 * 16];
  int lane = threadIdx.x & 31;
  int w    = threadIdx.x >> 5;
  int tile = blockIdx.x * 4 + w;
  bool active = tile < (NN / 16);
  int n  = lane & 15;
  int mb = (lane < 16) ? 0 : 8;
  if (active) {
    const float* At = agg + (long long)tile * 16 * CC;
    v16h a0 = frag_a(At, lane, 0, CC);
    v16h a1 = frag_a(At, lane, 1, CC);
    for (int nt = 0; nt < 4; ++nt) {
      v8f acc = {};
      acc = wmma_f16(a0, frag_bp(pwno, 2, nt, 0, lane), acc);
      acc = wmma_f16(a1, frag_bp(pwno, 2, nt, 1, lane), acc);
#pragma unroll
      for (int r = 0; r < 8; ++r) {
        int m = mb + r;
        long long row = (long long)tile * 16 + m;
        res[w][m * CC + nt * 16 + n] =
            node_states[row * CC + nt * 16 + n] + acc[r] + bno[nt * 16 + n];
      }
    }
  }
  __syncthreads();
  if (active && lane < 16) {
    const float* xr = res[w] + lane * CC;
    float s = 0.f, ss = 0.f;
    for (int j = 0; j < CC; ++j) { float x = xr[j]; s += x; ss += x * x; }
    float mean = s * (1.0f / CC);
    float rstd = rsqrtf(ss * (1.0f / CC) - mean * mean + LN_EPS);
    for (int j = 0; j < CC; ++j)
      pack_a_store(xln16[w], lane, j, (xr[j] - mean) * rstd * n2g[j] + n2b[j]);
  }
  __syncthreads();
  if (active) {
    v16h a0 = *(const v16h*)&xln16[w][(0 * 32 + lane) * 16];
    v16h a1 = *(const v16h*)&xln16[w][(1 * 32 + lane) * 16];
    for (int nt = 0; nt < 8; ++nt) {
      v8f acc = {};
      acc = wmma_f16(a0, frag_bp(pnf1w, 2, nt, 0, lane), acc);
      acc = wmma_f16(a1, frag_bp(pnf1w, 2, nt, 1, lane), acc);
#pragma unroll
      for (int r = 0; r < 8; ++r) {
        float vv = acc[r] + nf1b[nt * 16 + n];
        pack_a_store(h1p[w], mb + r, nt * 16 + n, vv > 0.f ? vv : 0.f);
      }
    }
  }
  __syncthreads();
  if (active) {
    v16h c0 = *(const v16h*)&h1p[w][(0 * 32 + lane) * 16];
    v16h c1 = *(const v16h*)&h1p[w][(1 * 32 + lane) * 16];
    v16h c2 = *(const v16h*)&h1p[w][(2 * 32 + lane) * 16];
    v16h c3 = *(const v16h*)&h1p[w][(3 * 32 + lane) * 16];
    for (int nt = 0; nt < 4; ++nt) {
      v8f acc = {};
      acc = wmma_f16(c0, frag_bp(pnf2w, 4, nt, 0, lane), acc);
      acc = wmma_f16(c1, frag_bp(pnf2w, 4, nt, 1, lane), acc);
      acc = wmma_f16(c2, frag_bp(pnf2w, 4, nt, 2, lane), acc);
      acc = wmma_f16(c3, frag_bp(pnf2w, 4, nt, 3, lane), acc);
#pragma unroll
      for (int r = 0; r < 8; ++r) {
        int m = mb + r;
        long long row = (long long)tile * 16 + m;
        out_node[row * CC + nt * 16 + n] =
            res[w][m * CC + nt * 16 + n] + acc[r] + nf2b[nt * 16 + n];
      }
    }
  }
}

// -------------------- edge finish: att@weo + residual, LN, FFN (WMMA)
__global__ __launch_bounds__(128) void k_edge_finish(
    const float* __restrict__ edge_states, const float* __restrict__ att,
    const float* __restrict__ weo, const float* __restrict__ beo,
    const float* __restrict__ e2g, const float* __restrict__ e2b,
    const half_t* __restrict__ pef1w, const float* __restrict__ ef1b,
    const half_t* __restrict__ pef2w, const float* __restrict__ ef2b,
    float* __restrict__ out_edge) {
  __shared__ alignas(32) float  res[4][16 * CC];
  __shared__ alignas(32) half_t xln16[4][2 * 32 * 16];
  __shared__ alignas(32) half_t h1p[4][4 * 32 * 16];
  int lane = threadIdx.x & 31;
  int w    = threadIdx.x >> 5;
  long long tile = (long long)blockIdx.x * 4 + w;  // E/16 tiles, exact
  int n  = lane & 15;
  int mb = (lane < 16) ? 0 : 8;
  // edge_update = attention @ weo (K=8, scalar MACs)
  for (int p = lane; p < 16 * CC; p += 32) {
    int el = p >> 6, nn2 = p & 63;
    long long e = tile * 16 + el;
    const float* ar = att + e * HH;
    float s = 0.f;
#pragma unroll
    for (int h = 0; h < HH; ++h) s += ar[h] * weo[h * CC + nn2];
    res[w][p] = edge_states[e * CC + nn2] + s + beo[nn2];
  }
  __syncthreads();
  if (lane < 16) {
    const float* xr = res[w] + lane * CC;
    float s = 0.f, ss = 0.f;
    for (int j = 0; j < CC; ++j) { float x = xr[j]; s += x; ss += x * x; }
    float mean = s * (1.0f / CC);
    float rstd = rsqrtf(ss * (1.0f / CC) - mean * mean + LN_EPS);
    for (int j = 0; j < CC; ++j)
      pack_a_store(xln16[w], lane, j, (xr[j] - mean) * rstd * e2g[j] + e2b[j]);
  }
  __syncthreads();
  {
    v16h a0 = *(const v16h*)&xln16[w][(0 * 32 + lane) * 16];
    v16h a1 = *(const v16h*)&xln16[w][(1 * 32 + lane) * 16];
    for (int nt = 0; nt < 8; ++nt) {
      v8f acc = {};
      acc = wmma_f16(a0, frag_bp(pef1w, 2, nt, 0, lane), acc);
      acc = wmma_f16(a1, frag_bp(pef1w, 2, nt, 1, lane), acc);
#pragma unroll
      for (int r = 0; r < 8; ++r) {
        float vv = acc[r] + ef1b[nt * 16 + n];
        pack_a_store(h1p[w], mb + r, nt * 16 + n, vv > 0.f ? vv : 0.f);
      }
    }
  }
  __syncthreads();
  {
    v16h c0 = *(const v16h*)&h1p[w][(0 * 32 + lane) * 16];
    v16h c1 = *(const v16h*)&h1p[w][(1 * 32 + lane) * 16];
    v16h c2 = *(const v16h*)&h1p[w][(2 * 32 + lane) * 16];
    v16h c3 = *(const v16h*)&h1p[w][(3 * 32 + lane) * 16];
    for (int nt = 0; nt < 4; ++nt) {
      v8f acc = {};
      acc = wmma_f16(c0, frag_bp(pef2w, 4, nt, 0, lane), acc);
      acc = wmma_f16(c1, frag_bp(pef2w, 4, nt, 1, lane), acc);
      acc = wmma_f16(c2, frag_bp(pef2w, 4, nt, 2, lane), acc);
      acc = wmma_f16(c3, frag_bp(pef2w, 4, nt, 3, lane), acc);
#pragma unroll
      for (int r = 0; r < 8; ++r) {
        int m = mb + r;
        long long row = tile * 16 + m;
        out_edge[row * CC + nt * 16 + n] =
            res[w][m * CC + nt * 16 + n] + acc[r] + ef2b[nt * 16 + n];
      }
    }
  }
}

extern "C" void kernel_launch(void* const* d_in, const int* in_sizes, int n_in,
                              void* d_out, int out_size, void* d_ws, size_t ws_size,
                              hipStream_t stream) {
  const float* node_states = (const float*)d_in[0];
  const float* edge_states = (const float*)d_in[1];
  const int*   edge_index  = (const int*)d_in[2];
  const float* wq  = (const float*)d_in[3];  const float* bq  = (const float*)d_in[4];
  const float* wk  = (const float*)d_in[5];  const float* bk  = (const float*)d_in[6];
  const float* wv  = (const float*)d_in[7];  const float* bv  = (const float*)d_in[8];
  const float* we  = (const float*)d_in[9];  const float* be  = (const float*)d_in[10];
  const float* wno = (const float*)d_in[11]; const float* bno = (const float*)d_in[12];
  const float* weo = (const float*)d_in[13]; const float* beo = (const float*)d_in[14];
  const float* n1g = (const float*)d_in[15]; const float* n1b = (const float*)d_in[16];
  const float* e1g = (const float*)d_in[17]; const float* e1b = (const float*)d_in[18];
  const float* n2g = (const float*)d_in[19]; const float* n2b = (const float*)d_in[20];
  const float* e2g = (const float*)d_in[21]; const float* e2b = (const float*)d_in[22];
  const float* nf1w = (const float*)d_in[23]; const float* nf1b = (const float*)d_in[24];
  const float* nf2w = (const float*)d_in[25]; const float* nf2b = (const float*)d_in[26];
  const float* ef1w = (const float*)d_in[27]; const float* ef1b = (const float*)d_in[28];
  const float* ef2w = (const float*)d_in[29]; const float* ef2b = (const float*)d_in[30];

  // workspace layout
  float* ws = (float*)d_ws;
  const size_t nc = (size_t)NN * CC;
  float* ni     = ws;
  float* qb     = ni + nc;
  float* kb     = qb + nc;
  float* vb     = kb + nc;
  float* agg    = vb + nc;
  float* scores = agg + nc;                                        // E*H
  unsigned int* segmax = (unsigned int*)(scores + (size_t)EE * HH); // N*H
  float* denom  = (float*)(segmax + (size_t)NN * HH);              // N*H
  half_t* pk0   = (half_t*)(denom + (size_t)NN * HH);
  half_t* pwq  = pk0;            half_t* pwk  = pwq  + 4096;
  half_t* pwv  = pwk  + 4096;    half_t* pwe  = pwv  + 4096;
  half_t* pwno = pwe  + 4096;    half_t* pnf1 = pwno + 4096;
  half_t* pnf2 = pnf1 + 8192;    half_t* pef1 = pnf2 + 8192;
  half_t* pef2 = pef1 + 8192;

  float* out_node = (float*)d_out;
  float* out_edge = out_node + nc;
  float* out_att  = out_edge + (size_t)EE * CC;

  // prepack weights into f16 B-fragment order
  k_pack_w<<<16, 256, 0, stream>>>(wq,   pwq,  64, 64);
  k_pack_w<<<16, 256, 0, stream>>>(wk,   pwk,  64, 64);
  k_pack_w<<<16, 256, 0, stream>>>(wv,   pwv,  64, 64);
  k_pack_w<<<16, 256, 0, stream>>>(we,   pwe,  64, 64);
  k_pack_w<<<16, 256, 0, stream>>>(wno,  pwno, 64, 64);
  k_pack_w<<<32, 256, 0, stream>>>(nf1w, pnf1, 64, 128);
  k_pack_w<<<32, 256, 0, stream>>>(nf2w, pnf2, 128, 64);
  k_pack_w<<<32, 256, 0, stream>>>(ef1w, pef1, 64, 128);
  k_pack_w<<<32, 256, 0, stream>>>(ef2w, pef2, 128, 64);

  long long initN = (long long)NN * CC;
  k_init<<<(unsigned)((initN + 255) / 256), 256, 0, stream>>>(agg, denom, segmax);

  int node_tiles = NN / 16;               // 6250
  int edge_tiles = EE / 16;               // 100000
  k_node_pre<<<(node_tiles + 3) / 4, 128, 0, stream>>>(
      node_states, n1g, n1b, pwq, bq, pwk, bk, pwv, bv, ni, qb, kb, vb);

  k_edge_score<<<edge_tiles / 4, 128, 0, stream>>>(
      edge_states, e1g, e1b, pwe, be, edge_index, qb, kb, scores, segmax);

  long long eh = (long long)EE * HH;
  k_exp_denom<<<(unsigned)((eh + 255) / 256), 256, 0, stream>>>(
      edge_index, scores, segmax, denom);

  k_attn_scatter<<<(unsigned)((eh + 255) / 256), 256, 0, stream>>>(
      edge_index, scores, denom, vb, ni, out_att, agg);

  k_node_finish<<<(node_tiles + 3) / 4, 128, 0, stream>>>(
      node_states, agg, pwno, bno, n2g, n2b, pnf1, nf1b, pnf2, nf2b, out_node);

  k_edge_finish<<<edge_tiles / 4, 128, 0, stream>>>(
      edge_states, out_att, weo, beo, e2g, e2b, pef1, ef1b, pef2, ef2b, out_edge);
}
